// ReferenceExperts_28767690949266
// MI455X (gfx1250) — matrix operand order repaired and checked by reading, not compiled, not run on hardware
//
#include <hip/hip_runtime.h>

// ---------------------------------------------------------------------------
// Fused MoE experts (gate_up GEMM + clamp/GLU + down GEMM + routed combine)
// for MI455X / gfx1250.  bf16 WMMA (v_wmma_f32_16x16x32_bf16) + async-to-LDS
// tile DMA (global_load_async_to_lds_b128, ASYNCcnt) with double buffering.
// ---------------------------------------------------------------------------

typedef __attribute__((ext_vector_type(8)))  __bf16 v8bf;
typedef __attribute__((ext_vector_type(16))) __bf16 v16bf;
typedef __attribute__((ext_vector_type(8)))  float  v8f;

#define T_TOK 8192
#define HDIM  1024
#define DDIM  1024
#define NEXP  8
#define LDSROW 40  // LDS row stride in bf16 elems: 80B rows, 16B aligned, bank-friendly

__device__ __forceinline__ v16bf frag_cat(v8bf lo, v8bf hi) {
  return __builtin_shufflevector(lo, hi, 0, 1, 2, 3, 4, 5, 6, 7,
                                 8, 9, 10, 11, 12, 13, 14, 15);
}

// A fragment 16x32 bf16: lane m = l&15; K = {0..7,16..23} (lanes 0-15),
// {8..15,24..31} (lanes 16-31).  LDS tile is [row][K] with K contiguous.
__device__ __forceinline__ v16bf load_fragA(const __bf16* s, int row0, int lane) {
  const __bf16* p = s + (row0 + (lane & 15)) * LDSROW + ((lane >> 4) << 3);
  return frag_cat(*(const v8bf*)p, *(const v8bf*)(p + 16));
}

// B fragment 32x16 bf16: lane n = l&15; K = {0..7,16..23} (lanes 0-15),
// {8..15,24..31} (lanes 16-31).
// LDS tile is [n][K] with K contiguous (weights pre-transposed to N-major).
__device__ __forceinline__ v16bf load_fragB(const __bf16* s, int row0, int lane) {
  const __bf16* p = s + (row0 + (lane & 15)) * LDSROW + ((lane >> 4) << 3);
  return frag_cat(*(const v8bf*)p, *(const v8bf*)(p + 16));
}

__device__ __forceinline__ v8f wmma_bf16(v16bf a, v16bf b, v8f c) {
  return __builtin_amdgcn_wmma_f32_16x16x32_bf16(false, a, false, b,
                                                 (short)0, c, false, false);
}

// --- CDNA5 async-to-LDS copy: 16B per lane, register-file bypass, ASYNCcnt --
__device__ __forceinline__ unsigned lds_off(const void* p) {
  // generic LDS pointer = {aperture_hi32, lds_offset_lo32}; async ops want the
  // 32-bit wave-relative LDS address.
  return (unsigned)(unsigned long long)p;
}

__device__ __forceinline__ void async_cp16(unsigned lds_addr, const void* gptr) {
  asm volatile("global_load_async_to_lds_b128 %0, %1, off"
               :
               : "v"(lds_addr), "v"((unsigned long long)gptr)
               : "memory");
}

template <int N>
__device__ __forceinline__ void wait_async() {
  asm volatile("s_wait_asynccnt %0" : : "i"(N) : "memory");
}

// ---------------------------------------------------------------------------
// Pre-pass: fp32 -> bf16 convert (hidden states)
// ---------------------------------------------------------------------------
__global__ __launch_bounds__(256) void k_cvt_bf16(const float* __restrict__ src,
                                                  unsigned short* __restrict__ dst_u,
                                                  size_t n) {
  __bf16* dst = (__bf16*)dst_u;
  size_t i = (size_t)blockIdx.x * blockDim.x + threadIdx.x;
  if (i < n) dst[i] = (__bf16)src[i];
}

// gate_up_proj [E][H][2D] f32  ->  [E][2][D][H] bf16 (gate plane, up plane)
__global__ __launch_bounds__(1024) void k_tr_wgu(const float* __restrict__ src,
                                                 unsigned short* __restrict__ dst_u) {
  __bf16* dst = (__bf16*)dst_u;
  __shared__ float tg[32][33];
  __shared__ float tu[32][33];
  const int e = blockIdx.z;
  const int d0 = blockIdx.x * 32;
  const int h0 = blockIdx.y * 32;
  const int tx = threadIdx.x, ty = threadIdx.y;
  const float2 v =
      *(const float2*)&src[((size_t)e * HDIM + h0 + ty) * (2 * DDIM) + 2 * (d0 + tx)];
  tg[ty][tx] = v.x;
  tu[ty][tx] = v.y;
  __syncthreads();
  dst[((size_t)(e * 2 + 0) * DDIM + d0 + ty) * HDIM + h0 + tx] = (__bf16)tg[tx][ty];
  dst[((size_t)(e * 2 + 1) * DDIM + d0 + ty) * HDIM + h0 + tx] = (__bf16)tu[tx][ty];
}

// down_proj [E][D][H] f32 -> [E][H][D] bf16
__global__ __launch_bounds__(1024) void k_tr_wd(const float* __restrict__ src,
                                                unsigned short* __restrict__ dst_u) {
  __bf16* dst = (__bf16*)dst_u;
  __shared__ float t[32][33];
  const int e = blockIdx.z;
  const int h0 = blockIdx.x * 32;
  const int d0 = blockIdx.y * 32;
  const int tx = threadIdx.x, ty = threadIdx.y;
  t[ty][tx] = src[((size_t)e * DDIM + d0 + ty) * HDIM + h0 + tx];
  __syncthreads();
  dst[((size_t)e * HDIM + h0 + ty) * DDIM + d0 + tx] = (__bf16)t[tx][ty];
}

// ---------------------------------------------------------------------------
// GEMM-1 + activation: act[e][t][d] = (clip(up)+1) * glu(min(gate,7))
// Block: 128 tokens x 64 act-cols; 8 wave32 in 4x2; 32x32 per wave; K-step 32.
// Double-buffered async-to-LDS pipeline (4 async ops / thread / stage).
// ---------------------------------------------------------------------------
__global__ __launch_bounds__(256) void k_gateup(
    const unsigned short* __restrict__ hs_u,   // [T][H] bf16
    const unsigned short* __restrict__ wgu_u,  // [E][2][D][H] bf16
    const float* __restrict__ bias,            // [E][2D] f32 interleaved
    unsigned short* __restrict__ act_u)        // [E][T][D] bf16
{
  const __bf16* hs = (const __bf16*)hs_u;
  const __bf16* wgu = (const __bf16*)wgu_u;
  __bf16* act = (__bf16*)act_u;

  __shared__ __align__(16) __bf16 sA[2][128 * LDSROW];
  __shared__ __align__(16) __bf16 sG[2][64 * LDSROW];
  __shared__ __align__(16) __bf16 sU[2][64 * LDSROW];

  const int tid = threadIdx.x;
  const int lane = tid & 31;
  const int wid = tid >> 5;
  const int wrow = wid >> 1;  // 0..3 -> 32-token offset
  const int wcol = wid & 1;   // 0..1 -> 32-col offset
  const int e = blockIdx.z;
  const int t0 = blockIdx.x * 128;
  const int d0 = blockIdx.y * 64;

  const __bf16* gW = wgu + ((size_t)(e * 2 + 0) * DDIM + d0) * HDIM;
  const __bf16* uW = wgu + ((size_t)(e * 2 + 1) * DDIM + d0) * HDIM;

  // per-thread 16B-chunk mapping: row = tid>>2 (and +64 for 2nd A chunk),
  // seg = tid&3 (8 bf16 per seg)
  const int crow = tid >> 2, cseg = tid & 3;
  const __bf16* gA0 = hs + (size_t)(t0 + crow) * HDIM + cseg * 8;
  const __bf16* gA1 = gA0 + (size_t)64 * HDIM;
  const __bf16* gGp = gW + (size_t)crow * HDIM + cseg * 8;
  const __bf16* gUp = uW + (size_t)crow * HDIM + cseg * 8;
  const int ldst = crow * LDSROW + cseg * 8;
  const unsigned dA0[2] = {lds_off(&sA[0][ldst]), lds_off(&sA[1][ldst])};
  const unsigned dA1[2] = {lds_off(&sA[0][ldst + 64 * LDSROW]),
                           lds_off(&sA[1][ldst + 64 * LDSROW])};
  const unsigned dG[2] = {lds_off(&sG[0][ldst]), lds_off(&sG[1][ldst])};
  const unsigned dU[2] = {lds_off(&sU[0][ldst]), lds_off(&sU[1][ldst])};

  v8f cg[2][2] = {};
  v8f cu[2][2] = {};

#define GU_ISSUE(K0, B)              \
  do {                               \
    async_cp16(dA0[B], gA0 + (K0));  \
    async_cp16(dA1[B], gA1 + (K0));  \
    async_cp16(dG[B], gGp + (K0));   \
    async_cp16(dU[B], gUp + (K0));   \
  } while (0)

#define GU_COMPUTE(B)                                        \
  do {                                                       \
    v16bf a0 = load_fragA(sA[B], wrow * 32, lane);           \
    v16bf a1 = load_fragA(sA[B], wrow * 32 + 16, lane);      \
    v16bf g0 = load_fragB(sG[B], wcol * 32, lane);           \
    v16bf g1 = load_fragB(sG[B], wcol * 32 + 16, lane);      \
    v16bf u0 = load_fragB(sU[B], wcol * 32, lane);           \
    v16bf u1 = load_fragB(sU[B], wcol * 32 + 16, lane);      \
    cg[0][0] = wmma_bf16(a0, g0, cg[0][0]);                  \
    cg[0][1] = wmma_bf16(a0, g1, cg[0][1]);                  \
    cg[1][0] = wmma_bf16(a1, g0, cg[1][0]);                  \
    cg[1][1] = wmma_bf16(a1, g1, cg[1][1]);                  \
    cu[0][0] = wmma_bf16(a0, u0, cu[0][0]);                  \
    cu[0][1] = wmma_bf16(a0, u1, cu[0][1]);                  \
    cu[1][0] = wmma_bf16(a1, u0, cu[1][0]);                  \
    cu[1][1] = wmma_bf16(a1, u1, cu[1][1]);                  \
  } while (0)

  GU_ISSUE(0, 0);
  for (int s = 0; s < 32; s += 2) {
    // stage s in buf0, stage s+1 in buf1
    GU_ISSUE((s + 1) * 32, 1);
    wait_async<4>();  // stage-s loads (oldest 4) landed
    __syncthreads();
    GU_COMPUTE(0);
    __syncthreads();  // all waves done reading buf0 -> safe to refill
    if (s + 2 < 32) {
      GU_ISSUE((s + 2) * 32, 0);
      wait_async<4>();
    } else {
      wait_async<0>();
    }
    __syncthreads();
    GU_COMPUTE(1);
    __syncthreads();
  }
#undef GU_ISSUE
#undef GU_COMPUTE

  // fused bias + clamp + GLU epilogue; C layout: n=lane&15, m=r+8*(lane>=16)
  const float* bG = bias + (size_t)e * 2 * DDIM;
#pragma unroll
  for (int mi = 0; mi < 2; ++mi)
#pragma unroll
    for (int ni = 0; ni < 2; ++ni) {
      const int dg = d0 + wcol * 32 + ni * 16 + (lane & 15);
      const float bgv = bG[2 * dg];
      const float buv = bG[2 * dg + 1];
#pragma unroll
      for (int r = 0; r < 8; ++r) {
        const int tl = t0 + wrow * 32 + mi * 16 + r + ((lane >> 4) << 3);
        float g = fminf(cg[mi][ni][r] + bgv, 7.0f);
        float u = fminf(fmaxf(cu[mi][ni][r] + buv, -7.0f), 7.0f);
        const float glu = g / (1.0f + __expf(-1.702f * g));
        act[((size_t)e * T_TOK + tl) * DDIM + dg] = (__bf16)((u + 1.0f) * glu);
      }
    }
}

// ---------------------------------------------------------------------------
// GEMM-2 + routed combine: out[t][h] = sum_e rw[t][e]*(act[e] @ Wd[e] + bd[e])
// Block: 128 tokens x 64 h; experts looped inside (no atomics).
// Double-buffered async-to-LDS pipeline (3 async ops / thread / stage).
// ---------------------------------------------------------------------------
__global__ __launch_bounds__(256) void k_down(
    const unsigned short* __restrict__ act_u,  // [E][T][D] bf16
    const unsigned short* __restrict__ wd_u,   // [E][H][D] bf16
    const float* __restrict__ dbias,           // [E][H]
    const float* __restrict__ rw,              // [T][E]
    float* __restrict__ out)                   // [T][H]
{
  const __bf16* act = (const __bf16*)act_u;
  const __bf16* wd = (const __bf16*)wd_u;

  __shared__ __align__(16) __bf16 sA[2][128 * LDSROW];
  __shared__ __align__(16) __bf16 sB[2][64 * LDSROW];
  __shared__ __align__(16) float sRW[128 * NEXP];

  const int tid = threadIdx.x;
  const int lane = tid & 31;
  const int wid = tid >> 5;
  const int wrow = wid >> 1;
  const int wcol = wid & 1;
  const int t0 = blockIdx.x * 128;
  const int h0 = blockIdx.y * 64;

  {  // stage routing weights once: 128 x 8 f32
    int row = tid >> 1, half = tid & 1;
    *(float4*)&sRW[row * NEXP + half * 4] =
        *(const float4*)&rw[(size_t)(t0 + row) * NEXP + half * 4];
  }

  const int crow = tid >> 2, cseg = tid & 3;
  const int ldst = crow * LDSROW + cseg * 8;
  const unsigned dA0[2] = {lds_off(&sA[0][ldst]), lds_off(&sA[1][ldst])};
  const unsigned dA1[2] = {lds_off(&sA[0][ldst + 64 * LDSROW]),
                           lds_off(&sA[1][ldst + 64 * LDSROW])};
  const unsigned dB[2] = {lds_off(&sB[0][ldst]), lds_off(&sB[1][ldst])};

  v8f acc[2][2] = {};

  for (int e = 0; e < NEXP; ++e) {
    const __bf16* gA0 =
        act + ((size_t)e * T_TOK + t0 + crow) * DDIM + cseg * 8;
    const __bf16* gA1 = gA0 + (size_t)64 * DDIM;
    const __bf16* gBp = wd + ((size_t)e * HDIM + h0 + crow) * DDIM + cseg * 8;

    v8f c[2][2] = {};

#define DN_ISSUE(K0, B)              \
  do {                               \
    async_cp16(dA0[B], gA0 + (K0));  \
    async_cp16(dA1[B], gA1 + (K0));  \
    async_cp16(dB[B], gBp + (K0));   \
  } while (0)

#define DN_COMPUTE(B)                                     \
  do {                                                    \
    v16bf a0 = load_fragA(sA[B], wrow * 32, lane);        \
    v16bf a1 = load_fragA(sA[B], wrow * 32 + 16, lane);   \
    v16bf b0 = load_fragB(sB[B], wcol * 32, lane);        \
    v16bf b1 = load_fragB(sB[B], wcol * 32 + 16, lane);   \
    c[0][0] = wmma_bf16(a0, b0, c[0][0]);                 \
    c[0][1] = wmma_bf16(a0, b1, c[0][1]);                 \
    c[1][0] = wmma_bf16(a1, b0, c[1][0]);                 \
    c[1][1] = wmma_bf16(a1, b1, c[1][1]);                 \
  } while (0)

    DN_ISSUE(0, 0);
    for (int s = 0; s < 32; s += 2) {
      DN_ISSUE((s + 1) * 32, 1);
      wait_async<3>();
      __syncthreads();
      DN_COMPUTE(0);
      __syncthreads();
      if (s + 2 < 32) {
        DN_ISSUE((s + 2) * 32, 0);
        wait_async<3>();
      } else {
        wait_async<0>();
      }
      __syncthreads();
      DN_COMPUTE(1);
      __syncthreads();
    }
#undef DN_ISSUE
#undef DN_COMPUTE

    const float* bD = dbias + (size_t)e * HDIM;
#pragma unroll
    for (int mi = 0; mi < 2; ++mi)
#pragma unroll
      for (int ni = 0; ni < 2; ++ni) {
        const int hg = h0 + wcol * 32 + ni * 16 + (lane & 15);
        const float bv = bD[hg];
#pragma unroll
        for (int r = 0; r < 8; ++r) {
          const int ml = wrow * 32 + mi * 16 + r + ((lane >> 4) << 3);
          acc[mi][ni][r] += sRW[ml * NEXP + e] * (c[mi][ni][r] + bv);
        }
      }
  }

#pragma unroll
  for (int mi = 0; mi < 2; ++mi)
#pragma unroll
    for (int ni = 0; ni < 2; ++ni) {
      const int hg = h0 + wcol * 32 + ni * 16 + (lane & 15);
#pragma unroll
      for (int r = 0; r < 8; ++r) {
        const int tl = t0 + wrow * 32 + mi * 16 + r + ((lane >> 4) << 3);
        out[(size_t)tl * HDIM + hg] = acc[mi][ni][r];
      }
    }
}

// ---------------------------------------------------------------------------
extern "C" void kernel_launch(void* const* d_in, const int* in_sizes, int n_in,
                              void* d_out, int out_size, void* d_ws, size_t ws_size,
                              hipStream_t stream) {
  const float* hs_f  = (const float*)d_in[0];  // [T][H]
  const float* rw    = (const float*)d_in[1];  // [T][E]
  const float* wgu_f = (const float*)d_in[2];  // [E][H][2D]
  const float* bgu   = (const float*)d_in[3];  // [E][2D]
  const float* wd_f  = (const float*)d_in[4];  // [E][D][H]
  const float* bd    = (const float*)d_in[5];  // [E][H]
  float* out = (float*)d_out;                  // [T][H] f32

  // workspace layout (all 16B aligned): ~201 MB total
  char* p = (char*)d_ws;
  unsigned short* hs_bf = (unsigned short*)p;
  p += (size_t)T_TOK * HDIM * 2;                       // 16 MB
  unsigned short* wgu_t = (unsigned short*)p;
  p += (size_t)NEXP * 2 * DDIM * HDIM * 2;             // 33.5 MB
  unsigned short* wd_t = (unsigned short*)p;
  p += (size_t)NEXP * HDIM * DDIM * 2;                 // 16.8 MB
  unsigned short* act_b = (unsigned short*)p;          // 134 MB

  const size_t n_hs = (size_t)T_TOK * HDIM;
  k_cvt_bf16<<<dim3((unsigned)(n_hs / 256)), 256, 0, stream>>>(hs_f, hs_bf, n_hs);

  dim3 b2(32, 32);
  k_tr_wgu<<<dim3(DDIM / 32, HDIM / 32, NEXP), b2, 0, stream>>>(wgu_f, wgu_t);
  k_tr_wd<<<dim3(HDIM / 32, DDIM / 32, NEXP), b2, 0, stream>>>(wd_f, wd_t);

  k_gateup<<<dim3(T_TOK / 128, DDIM / 64, NEXP), 256, 0, stream>>>(hs_bf, wgu_t, bgu, act_b);
  k_down<<<dim3(T_TOK / 128, HDIM / 64), 256, 0, stream>>>(act_b, wd_t, bd, rw, out);
}
